// Embedder_66022237274755
// MI455X (gfx1250) — matrix operand order repaired and checked by reading, not compiled, not run on hardware
//
#include <hip/hip_runtime.h>
#include <hip/hip_bf16.h>

// ---------------------------------------------------------------------------
// Types for CDNA5 WMMA bf16: A/B fragments are 16 bf16 per lane (8 VGPRs),
// C/D accumulator is 8 f32 per lane.
// ---------------------------------------------------------------------------
typedef __attribute__((ext_vector_type(16))) __bf16 v16bf;
typedef __attribute__((ext_vector_type(8)))  float  f32x8;

union Frag {
    v16bf v;
    uint4 q[2];
};

#define NB   100      // B * TOPK
#define TOPK 25
#define RES  128

// ---------------------------------------------------------------------------
// 1) Build masked inputs: x[100,128,128] fp32
// ---------------------------------------------------------------------------
__global__ void mask_kernel(const float* __restrict__ heat,
                            const float* __restrict__ dets,
                            float* __restrict__ x)
{
    int idx = blockIdx.x * blockDim.x + threadIdx.x;
    const int total = NB * RES * RES;
    if (idx >= total) return;
    int px  = idx & (RES - 1);
    int py  = (idx >> 7) & (RES - 1);
    int img = idx >> 14;
    int bi  = img / TOPK;
    const float* d = dets + img * 6;
    int left  = (int)floorf(d[0]);
    int upper = (int)floorf(d[1]);
    int right = (int)floorf(d[0] + d[2]);
    int lower = (int)floorf(d[1] + d[3]);
    float v = heat[bi * RES * RES + py * RES + px];
    bool in = (px >= left) && (px < right) && (py >= upper) && (py < lower);
    x[idx] = in ? v : 0.0f;
}

// ---------------------------------------------------------------------------
// 2) Stem: 7x7 s2 pad3 conv (1 -> 64 ch) + BN + ReLU -> NHWC bf16 [N,64,64,64]
// ---------------------------------------------------------------------------
__global__ void conv1_kernel(const float* __restrict__ x,
                             const float* __restrict__ w,   // [64][49] (OIHW, C=1)
                             const float* __restrict__ sc,
                             const float* __restrict__ bi,
                             __hip_bfloat16* __restrict__ out)
{
    int idx = blockIdx.x * blockDim.x + threadIdx.x;
    const int total = NB * 64 * 64 * 64;
    if (idx >= total) return;
    int c = idx & 63;
    int q = (idx >> 6) & 63;
    int p = (idx >> 12) & 63;
    int n = idx >> 18;
    const float* img = x + (size_t)n * RES * RES;
    float acc = 0.0f;
    #pragma unroll
    for (int r = 0; r < 7; ++r) {
        int ih = p * 2 - 3 + r;
        if ((unsigned)ih >= (unsigned)RES) continue;
        #pragma unroll
        for (int s = 0; s < 7; ++s) {
            int iw = q * 2 - 3 + s;
            if ((unsigned)iw >= (unsigned)RES) continue;
            acc += img[ih * RES + iw] * w[c * 49 + r * 7 + s];
        }
    }
    float v = acc * sc[c] + bi[c];
    v = v > 0.0f ? v : 0.0f;
    out[idx] = __float2bfloat16(v);
}

// ---------------------------------------------------------------------------
// 3) 3x3 s2 pad1 maxpool on NHWC bf16: [N,64,64,64] -> [N,32,32,64]
// ---------------------------------------------------------------------------
__global__ void maxpool_kernel(const __hip_bfloat16* __restrict__ in,
                               __hip_bfloat16* __restrict__ out)
{
    int idx = blockIdx.x * blockDim.x + threadIdx.x;
    const int total = NB * 32 * 32 * 64;
    if (idx >= total) return;
    int c = idx & 63;
    int q = (idx >> 6) & 31;
    int p = (idx >> 11) & 31;
    int n = idx >> 16;
    float m = -INFINITY;
    for (int dy = 0; dy < 3; ++dy) {
        int ih = 2 * p - 1 + dy;
        if ((unsigned)ih >= 64u) continue;
        for (int dx = 0; dx < 3; ++dx) {
            int iw = 2 * q - 1 + dx;
            if ((unsigned)iw >= 64u) continue;
            float v = __bfloat162float(in[(((size_t)n * 64 + ih) * 64 + iw) * 64 + c]);
            m = v > m ? v : m;
        }
    }
    out[idx] = __float2bfloat16(m);
}

// ---------------------------------------------------------------------------
// 4) Weight reorder: fp32 OIHW -> bf16 [O][ (r*S+s)*C + c ]  (c innermost)
// ---------------------------------------------------------------------------
__global__ void reorder_w_kernel(const float* __restrict__ src,
                                 __hip_bfloat16* __restrict__ dst,
                                 int O, int C, int R, int S)
{
    int idx = blockIdx.x * blockDim.x + threadIdx.x;
    int RSC = R * S * C;
    if (idx >= O * RSC) return;
    int o   = idx / RSC;
    int rem = idx - o * RSC;
    int c   = rem % C;
    int rs  = rem / C;
    int r   = rs / S;
    int s   = rs - r * S;
    dst[idx] = __float2bfloat16(src[(((size_t)o * C + c) * R + r) * S + s]);
}

// ---------------------------------------------------------------------------
// 5) Implicit-GEMM convolution with WMMA bf16.
//    One wave (32 threads) computes a 32(M=output pixels) x 64(N=out chans)
//    tile: 2 A-fragments x 4 B-fragments = 8 v_wmma per K-step of 32.
//    K-dim = (r*S+s)*C + c, c innermost; C % 32 == 0, so each 32-wide K
//    chunk is one spatial tap and a contiguous NHWC channel run.
//    R,S are template params -> tap loops fully unrolled, no integer division
//    anywhere (P=Q are powers of two, decoded via shifts).
//    Epilogue fuses BN (scale,bias), optional residual add (bf16), ReLU.
// ---------------------------------------------------------------------------
template<int R, int S>
__global__ __launch_bounds__(32)
void conv_wmma_kernel(const __hip_bfloat16* __restrict__ act,    // NHWC [N,H,W,C]
                      const __hip_bfloat16* __restrict__ wt,     // [Kout][R*S*C]
                      const float* __restrict__ scale,
                      const float* __restrict__ bias,
                      const __hip_bfloat16* __restrict__ resid,  // NHWC [N,P,Q,Kout] or null
                      __hip_bfloat16* __restrict__ out,          // NHWC [N,P,Q,Kout]
                      int H, int W, int C, int Kout,
                      int stride, int pad, int Q, int qlog, int relu)
{
    const int lane = threadIdx.x;
    const int half = (lane >> 4) & 1;
    const int l15  = lane & 15;

    const int mtile = blockIdx.x;   // M/32
    const int ntile = blockIdx.y;   // Kout/64

    const int RSC = R * S * C;

    // Per-lane pixel for the two A fragments (rows l15 and l15+16 of the tile).
    int ih0[2], iwq[2];
    size_t rowbase[2];
    #pragma unroll
    for (int f = 0; f < 2; ++f) {
        int mg = mtile * 32 + f * 16 + l15;
        int q_ = mg & (Q - 1);
        int p_ = (mg >> qlog) & (Q - 1);     // P == Q (square maps)
        int n_ = mg >> (2 * qlog);
        ih0[f] = p_ * stride - pad;
        iwq[f] = q_ * stride - pad;
        rowbase[f] = (size_t)n_ * H;
    }

    f32x8 acc[2][4] = {};

    const __hip_bfloat16* w0 = wt + (size_t)(ntile * 64 + 0 * 16 + l15) * RSC;
    const __hip_bfloat16* w1 = wt + (size_t)(ntile * 64 + 1 * 16 + l15) * RSC;
    const __hip_bfloat16* w2 = wt + (size_t)(ntile * 64 + 2 * 16 + l15) * RSC;
    const __hip_bfloat16* w3 = wt + (size_t)(ntile * 64 + 3 * 16 + l15) * RSC;
    const int kboff = half * 16;   // B: lane half owns 16 contiguous K values
    const int aoff  = half * 16;   // A: lane half owns 16 contiguous K values

    int kb = kboff;                // running B K-offset (advances 32 per chunk)

    #pragma unroll
    for (int r = 0; r < R; ++r) {
        #pragma unroll
        for (int s = 0; s < S; ++s) {
            // Per-tap activation base + validity (per lane, per fragment).
            const __hip_bfloat16* ap[2];
            bool ok[2];
            #pragma unroll
            for (int f = 0; f < 2; ++f) {
                int ih = ih0[f] + r;
                int iw = iwq[f] + s;
                ok[f] = (unsigned)ih < (unsigned)H && (unsigned)iw < (unsigned)W;
                ap[f] = act + ((rowbase[f] + ih) * W + iw) * C + aoff;
            }

            for (int c0 = 0; c0 < C; c0 += 32, kb += 32) {
                Frag a0, a1;
                if (ok[0]) {
                    a0.q[0] = *(const uint4*)(ap[0] + c0);
                    a0.q[1] = *(const uint4*)(ap[0] + c0 + 8);
                } else {
                    a0.q[0] = make_uint4(0, 0, 0, 0);
                    a0.q[1] = make_uint4(0, 0, 0, 0);
                }
                if (ok[1]) {
                    a1.q[0] = *(const uint4*)(ap[1] + c0);
                    a1.q[1] = *(const uint4*)(ap[1] + c0 + 8);
                } else {
                    a1.q[0] = make_uint4(0, 0, 0, 0);
                    a1.q[1] = make_uint4(0, 0, 0, 0);
                }

                Frag b0, b1, b2, b3;
                b0.q[0] = *(const uint4*)(w0 + kb); b0.q[1] = *(const uint4*)(w0 + kb + 8);
                b1.q[0] = *(const uint4*)(w1 + kb); b1.q[1] = *(const uint4*)(w1 + kb + 8);
                b2.q[0] = *(const uint4*)(w2 + kb); b2.q[1] = *(const uint4*)(w2 + kb + 8);
                b3.q[0] = *(const uint4*)(w3 + kb); b3.q[1] = *(const uint4*)(w3 + kb + 8);

                acc[0][0] = __builtin_amdgcn_wmma_f32_16x16x32_bf16(false, a0.v, false, b0.v, (short)0, acc[0][0], false, false);
                acc[0][1] = __builtin_amdgcn_wmma_f32_16x16x32_bf16(false, a0.v, false, b1.v, (short)0, acc[0][1], false, false);
                acc[0][2] = __builtin_amdgcn_wmma_f32_16x16x32_bf16(false, a0.v, false, b2.v, (short)0, acc[0][2], false, false);
                acc[0][3] = __builtin_amdgcn_wmma_f32_16x16x32_bf16(false, a0.v, false, b3.v, (short)0, acc[0][3], false, false);
                acc[1][0] = __builtin_amdgcn_wmma_f32_16x16x32_bf16(false, a1.v, false, b0.v, (short)0, acc[1][0], false, false);
                acc[1][1] = __builtin_amdgcn_wmma_f32_16x16x32_bf16(false, a1.v, false, b1.v, (short)0, acc[1][1], false, false);
                acc[1][2] = __builtin_amdgcn_wmma_f32_16x16x32_bf16(false, a1.v, false, b2.v, (short)0, acc[1][2], false, false);
                acc[1][3] = __builtin_amdgcn_wmma_f32_16x16x32_bf16(false, a1.v, false, b3.v, (short)0, acc[1][3], false, false);
            }
        }
    }

    // Epilogue: D layout -> VGPR e holds row (e + half*8), col = l15.
    #pragma unroll
    for (int f = 0; f < 2; ++f) {
        const int mbase = mtile * 32 + f * 16 + half * 8;
        #pragma unroll
        for (int sub = 0; sub < 4; ++sub) {
            int cg = ntile * 64 + sub * 16 + l15;
            float sc = scale[cg];
            float bi = bias[cg];
            f32x8 a = acc[f][sub];
            #pragma unroll
            for (int e = 0; e < 8; ++e) {
                size_t oidx = (size_t)(mbase + e) * Kout + cg;
                float v = a[e] * sc + bi;
                if (resid) v += __bfloat162float(resid[oidx]);
                if (relu)  v = v > 0.0f ? v : 0.0f;
                out[oidx] = __float2bfloat16(v);
            }
        }
    }
}

// ---------------------------------------------------------------------------
// 6) Global average pool: [N,4,4,2048] bf16 -> [N,2048] f32
// ---------------------------------------------------------------------------
__global__ void avgpool_kernel(const __hip_bfloat16* __restrict__ in,
                               float* __restrict__ out)
{
    int idx = blockIdx.x * blockDim.x + threadIdx.x;
    const int total = NB * 2048;
    if (idx >= total) return;
    int c = idx & 2047;
    int n = idx >> 11;
    float s = 0.0f;
    #pragma unroll
    for (int i = 0; i < 16; ++i)
        s += __bfloat162float(in[((size_t)n * 16 + i) * 2048 + c]);
    out[idx] = s * (1.0f / 16.0f);
}

// ---------------------------------------------------------------------------
// 7) FC: out[n,t] = dot(pooled[n,:], fc_w[t,:]) + fc_b[t]   (fp32; tiny)
// ---------------------------------------------------------------------------
__global__ void fc_kernel(const float* __restrict__ pooled,
                          const float* __restrict__ w,   // [512,2048]
                          const float* __restrict__ b,
                          float* __restrict__ out)
{
    int n = blockIdx.x;
    int t = threadIdx.x;   // 512
    const float* pv = pooled + (size_t)n * 2048;
    const float* wr = w + (size_t)t * 2048;
    float s = 0.0f;
    for (int i = 0; i < 2048; ++i) s += pv[i] * wr[i];
    out[(size_t)n * 512 + t] = s + b[t];
}

// ---------------------------------------------------------------------------
// 8) L2 normalize each 512-vector -> d_out fp32
// ---------------------------------------------------------------------------
__global__ void norm_kernel(const float* __restrict__ fcout,
                            float* __restrict__ out)
{
    __shared__ float sh[512];
    int n = blockIdx.x;
    int t = threadIdx.x;
    float v = fcout[(size_t)n * 512 + t];
    sh[t] = v * v;
    __syncthreads();
    for (int s = 256; s > 0; s >>= 1) {
        if (t < s) sh[t] += sh[t + s];
        __syncthreads();
    }
    out[(size_t)n * 512 + t] = v * rsqrtf(sh[0]);
}

// ---------------------------------------------------------------------------
// Host side
// ---------------------------------------------------------------------------
extern "C" void kernel_launch(void* const* d_in, const int* in_sizes, int n_in,
                              void* d_out, int out_size, void* d_ws, size_t ws_size,
                              hipStream_t stream)
{
    (void)in_sizes; (void)n_in; (void)out_size; (void)ws_size;

    const float* heat = (const float*)d_in[0];
    const float* dets = (const float*)d_in[1];
    int ci = 2;
    auto F = [&](void) { return (const float*)d_in[ci++]; };

    const float* conv1_w = F();
    const float* bn1_s   = F();
    const float* bn1_b   = F();

    // ---- workspace carve ----
    char* ws = (char*)d_ws;
    size_t off = 0;
    auto carve = [&](size_t bytes) -> void* {
        void* p = ws + off;
        off += (bytes + 255) & ~(size_t)255;
        return p;
    };
    __hip_bfloat16* bufA = (__hip_bfloat16*)carve(56ull << 20);
    __hip_bfloat16* bufB = (__hip_bfloat16*)carve(56ull << 20);
    __hip_bfloat16* bufC = (__hip_bfloat16*)carve(28ull << 20);
    float* x       = (float*)carve((size_t)NB * RES * RES * 4);
    float* pooled  = (float*)carve((size_t)NB * 2048 * 4);
    float* fcout   = (float*)carve((size_t)NB * 512 * 4);
    __hip_bfloat16* wreg = (__hip_bfloat16*)carve(56ull << 20);

    size_t wcur = 0;
    auto convert_w = [&](const float* src, int O, int C, int R, int S)
        -> const __hip_bfloat16* {
        __hip_bfloat16* dst = wreg + wcur;
        size_t total = (size_t)O * C * R * S;
        wcur += (total + 127) & ~(size_t)127;
        reorder_w_kernel<<<(int)((total + 255) / 256), 256, 0, stream>>>(
            src, dst, O, C, R, S);
        return dst;
    };

    auto conv = [&](const __hip_bfloat16* act, const __hip_bfloat16* wt,
                    const float* sc, const float* bi,
                    const __hip_bfloat16* resid, __hip_bfloat16* out,
                    int H, int W, int C, int Kout,
                    int R, int S, int stride, int pad, int relu) {
        int P = (H + 2 * pad - R) / stride + 1;
        int Q = (W + 2 * pad - S) / stride + 1;
        int qlog = (Q == 64) ? 6 : (Q == 32) ? 5 : (Q == 16) ? 4
                 : (Q == 8) ? 3 : 2;
        int M = NB * P * Q;                 // always a multiple of 32
        dim3 grid(M / 32, Kout / 64);
        if (R == 1)
            conv_wmma_kernel<1, 1><<<grid, 32, 0, stream>>>(
                act, wt, sc, bi, resid, out, H, W, C, Kout, stride, pad,
                Q, qlog, relu);
        else
            conv_wmma_kernel<3, 3><<<grid, 32, 0, stream>>>(
                act, wt, sc, bi, resid, out, H, W, C, Kout, stride, pad,
                Q, qlog, relu);
    };

    // ---- stem ----
    {
        int total = NB * RES * RES;
        mask_kernel<<<(total + 255) / 256, 256, 0, stream>>>(heat, dets, x);
        int t2 = NB * 64 * 64 * 64;
        conv1_kernel<<<(t2 + 255) / 256, 256, 0, stream>>>(x, conv1_w, bn1_s,
                                                           bn1_b, bufB);
        int t3 = NB * 32 * 32 * 64;
        maxpool_kernel<<<(t3 + 255) / 256, 256, 0, stream>>>(bufB, bufA);
    }

    // ---- stages ----
    static const int STG_W[4] = {64, 128, 256, 512};
    static const int STG_N[4] = {3, 4, 6, 3};
    static const int STG_S[4] = {1, 2, 2, 2};

    __hip_bfloat16* pA = bufA;   // current block input (and identity)
    __hip_bfloat16* pB = bufB;
    __hip_bfloat16* pC = bufC;
    int H = 32, W = 32, cin = 64;

    for (int si = 0; si < 4; ++si) {
        int wch = STG_W[si];
        int cout = 4 * wch;
        for (int bi2 = 0; bi2 < STG_N[si]; ++bi2) {
            int stride = (bi2 == 0) ? STG_S[si] : 1;
            bool ds = (bi2 == 0);

            // params in insertion order: w1,w2,w3,s1,b1,s2,b2,s3,b3[,wd,sd,bd]
            const float* w1 = F(); const float* w2 = F(); const float* w3 = F();
            const float* s1 = F(); const float* b1 = F();
            const float* s2 = F(); const float* b2 = F();
            const float* s3 = F(); const float* b3 = F();
            const float *wd = nullptr, *sd = nullptr, *bd = nullptr;
            if (ds) { wd = F(); sd = F(); bd = F(); }

            const __hip_bfloat16* W1 = convert_w(w1, wch, cin, 1, 1);
            const __hip_bfloat16* W2 = convert_w(w2, wch, wch, 3, 3);
            const __hip_bfloat16* W3 = convert_w(w3, cout, wch, 1, 1);
            const __hip_bfloat16* Wd = ds ? convert_w(wd, cout, cin, 1, 1) : nullptr;

            int P = (H - 1) / stride + 1;
            int Q = (W - 1) / stride + 1;

            // t1 = relu(bn(conv1x1(X)))           : pA -> pB   [H,W,wch]
            conv(pA, W1, s1, b1, nullptr, pB, H, W, cin, wch, 1, 1, 1, 0, 1);
            // t2 = relu(bn(conv3x3_s(t1)))        : pB -> pC   [P,Q,wch]
            conv(pB, W2, s2, b2, nullptr, pC, H, W, wch, wch, 3, 3, stride, 1, 1);
            if (ds) {
                // idt = bn(conv1x1_s(X))          : pA -> pB   [P,Q,cout]
                conv(pA, Wd, sd, bd, nullptr, pB, H, W, cin, cout, 1, 1, stride, 0, 0);
                // out = relu(bn(conv1x1(t2))+idt) : pC -> pA
                conv(pC, W3, s3, b3, pB, pA, P, Q, wch, cout, 1, 1, 1, 0, 1);
            } else {
                // out = relu(bn(conv1x1(t2))+X)   : pC -> pB
                conv(pC, W3, s3, b3, pA, pB, P, Q, wch, cout, 1, 1, 1, 0, 1);
                __hip_bfloat16* t = pA; pA = pB; pB = t;
            }
            H = P; W = Q; cin = cout;
        }
    }

    const float* fc_w = F();
    const float* fc_b = F();

    // ---- head ----
    {
        int t1 = NB * 2048;
        avgpool_kernel<<<(t1 + 255) / 256, 256, 0, stream>>>(pA, pooled);
        fc_kernel<<<NB, 512, 0, stream>>>(pooled, fc_w, fc_b, fcout);
        norm_kernel<<<NB, 512, 0, stream>>>(fcout, (float*)d_out);
    }
}
